// GNNModel_22574348108223
// MI455X (gfx1250) — compile-verified, matrix operand order checked by reading, not run on hardware
//
#include <hip/hip_runtime.h>

// ---------------- problem constants ----------------
#define G_    4096
#define NPG_  32
#define EPG_  128          // edges per graph
#define E_    524288
#define D1_   256
#define FF_   128
#define HP_   264          // h row pitch (bf16), 528B = 16B-aligned rows
#define TP_   40           // t1^T row pitch
#define AP_   40           // adjacency row pitch
#define NEGV  (-1e30f)

typedef __attribute__((ext_vector_type(16))) __bf16 v16bf;
typedef __attribute__((ext_vector_type(8)))  __bf16 v8bf;
typedef __attribute__((ext_vector_type(8)))  float  v8f;

__device__ inline v8f wmma_bf16(v16bf a, v16bf b, v8f c) {
  return __builtin_amdgcn_wmma_f32_16x16x32_bf16(false, a, false, b, (short)0, c,
                                                 false, false);
}

__device__ inline v16bf cat8(v8bf lo, v8bf hi) {
  union { v16bf v; v8bf h[2]; } u;
  u.h[0] = lo; u.h[1] = hi;
  return u.v;
}

// A-style 16x32 bf16 fragment from row-major LDS tile.
// Lane L<16: row=r0+L, K chunks {0..7} and {16..23}; lanes 16..31: K {8..15},{24..31}.
__device__ inline v16bf frag_rowmajor(const __bf16* base, int pitch, int r0, int k0,
                                      int lane) {
  int row = r0 + (lane & 15);
  int off = (lane >> 4) << 3;
  const __bf16* p = base + row * pitch + k0 + off;
  v8bf lo = *(const v8bf*)p;
  v8bf hi = *(const v8bf*)(p + 16);
  return cat8(lo, hi);
}

// B fragment from pre-packed weights: 512 bf16 per (ktile,ntile), lane-major.
__device__ inline v16bf frag_packed(const __bf16* pk, int kt, int nt, int ntiles,
                                    int lane) {
  return *(const v16bf*)(pk + (((kt * ntiles + nt) << 9) + (lane << 4)));
}

// C/D 16x16 f32 frag: lanes 0-15 rows 0..7 (vgpr r -> M=r), lanes 16-31 rows 8..15.
__device__ inline void store_c(__bf16* base, int pitch, int m0, int n0, int lane,
                               v8f c) {
  int col = n0 + (lane & 15);
  int rb  = m0 + ((lane >> 4) << 3);
#pragma unroll
  for (int r = 0; r < 8; ++r) base[(rb + r) * pitch + col] = (__bf16)c[r];
}

__device__ inline void store_cT(__bf16* base, int pitch, int m0, int n0, int lane,
                                v8f c) {
  int col = n0 + (lane & 15);
  int rb  = m0 + ((lane >> 4) << 3);
#pragma unroll
  for (int r = 0; r < 8; ++r) base[col * pitch + rb + r] = (__bf16)c[r];
}

// ---------------- one SAGE + TopK + GMP layer ----------------
template <int KT>
__device__ void layer_step(int t, int lane, int wave,
                           const __bf16* Wl, const __bf16* Wr,
                           const float* bl, const float* pn,
                           __bf16* sh, __bf16* st1, __bf16* sadj,
                           unsigned* scnt, unsigned* sdeg, float* ssc,
                           float* sscale, int* salive, int* skeep,
                           const unsigned char* sS, const unsigned char* sD,
                           float& greg) {
  __builtin_prefetch(Wl, 0, 1);
  __builtin_prefetch(Wr, 0, 1);

  // ---- build normalized adjacency (em[e] = alive[src]*alive[dst]) ----
  for (int j = t; j < NPG_ * NPG_; j += 128) scnt[j] = 0u;
  if (t < NPG_) { sdeg[t] = 0u; ssc[t] = 0.f; }
  __syncthreads();
  {
    int ls = sS[t], ld = sD[t];
    if (salive[ls] && salive[ld]) {
      atomicAdd(&scnt[ld * NPG_ + ls], 1u);
      atomicAdd(&sdeg[ld], 1u);
    }
  }
  __syncthreads();
  for (int j = t; j < NPG_ * NPG_; j += 128) {
    int i = j >> 5;
    float dv = fmaxf((float)sdeg[i], 1.f);
    sadj[i * AP_ + (j & 31)] = (__bf16)((float)scnt[j] / dv);
  }
  __syncthreads();

  // ---- dual GEMM: t1 = h@Wl, t2 = h@Wr  (wave w owns cols [32w,32w+32)) ----
  const int nt0 = wave * 2, nt1 = wave * 2 + 1;
  v8f t1a00 = {}, t1a01 = {}, t1a10 = {}, t1a11 = {};
  v8f t2a00 = {}, t2a01 = {}, t2a10 = {}, t2a11 = {};
#pragma unroll
  for (int kt = 0; kt < KT; ++kt) {
    v16bf a0 = frag_rowmajor(sh, HP_, 0, kt * 32, lane);
    v16bf a1 = frag_rowmajor(sh, HP_, 16, kt * 32, lane);
    v16bf bL0 = frag_packed(Wl, kt, nt0, FF_ / 16, lane);
    v16bf bL1 = frag_packed(Wl, kt, nt1, FF_ / 16, lane);
    v16bf bR0 = frag_packed(Wr, kt, nt0, FF_ / 16, lane);
    v16bf bR1 = frag_packed(Wr, kt, nt1, FF_ / 16, lane);
    t1a00 = wmma_bf16(a0, bL0, t1a00);
    t1a01 = wmma_bf16(a0, bL1, t1a01);
    t1a10 = wmma_bf16(a1, bL0, t1a10);
    t1a11 = wmma_bf16(a1, bL1, t1a11);
    t2a00 = wmma_bf16(a0, bR0, t2a00);
    t2a01 = wmma_bf16(a0, bR1, t2a01);
    t2a10 = wmma_bf16(a1, bR0, t2a10);
    t2a11 = wmma_bf16(a1, bR1, t2a11);
  }
  // stage t1 transposed (B-fragment-friendly) in LDS
  store_cT(st1, TP_, 0, nt0 * 16, lane, t1a00);
  store_cT(st1, TP_, 0, nt1 * 16, lane, t1a01);
  store_cT(st1, TP_, 16, nt0 * 16, lane, t1a10);
  store_cT(st1, TP_, 16, nt1 * 16, lane, t1a11);
  __syncthreads();

  // ---- out = Adj @ t1 + t2 (seed C with t2 accumulators), K = 32 ----
  v16bf aa0 = frag_rowmajor(sadj, AP_, 0, 0, lane);
  v16bf aa1 = frag_rowmajor(sadj, AP_, 16, 0, lane);
  v16bf tb0 = frag_rowmajor(st1, TP_, nt0 * 16, 0, lane);
  v16bf tb1 = frag_rowmajor(st1, TP_, nt1 * 16, 0, lane);
  v8f o00 = wmma_bf16(aa0, tb0, t2a00);
  v8f o01 = wmma_bf16(aa0, tb1, t2a01);
  v8f o10 = wmma_bf16(aa1, tb0, t2a10);
  v8f o11 = wmma_bf16(aa1, tb1, t2a11);

  // bias + ReLU, write new h (pre-topk-scale) into sh
  float bv0 = bl[nt0 * 16 + (lane & 15)];
  float bv1 = bl[nt1 * 16 + (lane & 15)];
#pragma unroll
  for (int r = 0; r < 8; ++r) {
    o00[r] = fmaxf(o00[r] + bv0, 0.f);
    o01[r] = fmaxf(o01[r] + bv1, 0.f);
    o10[r] = fmaxf(o10[r] + bv0, 0.f);
    o11[r] = fmaxf(o11[r] + bv1, 0.f);
  }
  store_c(sh, HP_, 0, nt0 * 16, lane, o00);
  store_c(sh, HP_, 0, nt1 * 16, lane, o01);
  store_c(sh, HP_, 16, nt0 * 16, lane, o10);
  store_c(sh, HP_, 16, nt1 * 16, lane, o11);
  __syncthreads();

  // ---- TopK: score = h . (p/||p||), rank within graph, keep top ceil(0.8*cnt) ----
  {
    int node = t >> 2, part = t & 3;
    float s = 0.f;
    int c0 = part * 32;
    for (int c = c0; c < c0 + 32; ++c) s += (float)sh[node * HP_ + c] * pn[c];
    atomicAdd(&ssc[node], s);
  }
  __syncthreads();
  if (t < NPG_) ssc[t] = salive[t] ? ssc[t] : NEGV;
  __syncthreads();
  if (t < NPG_) {
    float si = ssc[t];
    int rank = 0, cnt = 0;
    for (int j = 0; j < NPG_; ++j) {
      float sj = ssc[j];
      rank += (sj > si) || (sj == si && j < t);
      cnt += salive[j];
    }
    float kf = ceilf(0.8f * (float)cnt);
    int kp = ((float)rank < kf) && salive[t];
    skeep[t] = kp;
    sscale[t] = kp ? tanhf(si) : 0.f;
  }
  __syncthreads();
  // scale h (thread t owns column t), update alive
  for (int i = 0; i < NPG_; ++i) {
    float v = (float)sh[i * HP_ + t] * sscale[i];
    sh[i * HP_ + t] = (__bf16)v;
  }
  if (t < NPG_) salive[t] = skeep[t];
  __syncthreads();
  // ---- global mean pool over kept nodes ----
  {
    int cn = 0; float s = 0.f;
    for (int i = 0; i < NPG_; ++i) {
      if (skeep[i]) { cn++; s += (float)sh[i * HP_ + t]; }
    }
    greg += s / fmaxf((float)cn, 1.f);
  }
  __syncthreads();
}

// ---------------- main per-graph kernel ----------------
__global__ __launch_bounds__(128) void gnn_graph_kernel(
    const int* __restrict__ x, const int* __restrict__ src,
    const int* __restrict__ dst, const float* __restrict__ emb,
    const __bf16* __restrict__ pw, const float* __restrict__ b1,
    const float* __restrict__ b2, const float* __restrict__ b3,
    const float* __restrict__ pn, float* __restrict__ gout) {
  const int g = blockIdx.x;
  const int t = threadIdx.x;
  const int lane = t & 31;
  const int wave = t >> 5;

  __shared__ __align__(16) __bf16 sh[NPG_ * HP_];
  __shared__ __align__(16) __bf16 st1[FF_ * TP_];
  __shared__ __align__(16) __bf16 sadj[NPG_ * AP_];
  __shared__ unsigned scnt[NPG_ * NPG_];
  __shared__ unsigned sdeg[NPG_];
  __shared__ float ssc[NPG_];
  __shared__ float sscale[NPG_];
  __shared__ int salive[NPG_];
  __shared__ int skeep[NPG_];
  __shared__ int sX[NPG_];
  __shared__ unsigned char sS[EPG_], sD[EPG_];

  if (t < NPG_) { sX[t] = x[g * NPG_ + t]; salive[t] = 1; }
  sS[t] = (unsigned char)(src[g * EPG_ + t] & (NPG_ - 1));
  sD[t] = (unsigned char)(dst[g * EPG_ + t] & (NPG_ - 1));
  __syncthreads();

  // gather item embeddings -> bf16 LDS [32 x 256]
  for (int j = t; j < NPG_ * D1_; j += 128) {
    int row = j >> 8, col = j & 255;
    float v = emb[(long long)sX[row] * D1_ + col];
    sh[row * HP_ + col] = (__bf16)v;
  }
  float greg = 0.f;
  __syncthreads();

  layer_step<8>(t, lane, wave, pw, pw + 32768, b1, pn, sh, st1, sadj, scnt,
                sdeg, ssc, sscale, salive, skeep, sS, sD, greg);
  layer_step<4>(t, lane, wave, pw + 65536, pw + 81920, b2, pn + 128, sh, st1,
                sadj, scnt, sdeg, ssc, sscale, salive, skeep, sS, sD, greg);
  layer_step<4>(t, lane, wave, pw + 98304, pw + 114688, b3, pn + 256, sh, st1,
                sadj, scnt, sdeg, ssc, sscale, salive, skeep, sS, sD, greg);

  gout[g * FF_ + t] = greg;
}

// ---------------- weight packing: row-major f32 -> WMMA-B bf16 fragments ----
__global__ __launch_bounds__(256) void pack_weight(const float* __restrict__ W,
                                                   __bf16* __restrict__ out,
                                                   int K, int N) {
  int i = blockIdx.x * 256 + threadIdx.x;
  if (i >= K * N) return;
  int tile = i >> 9, r = i & 511;
  int lane = r >> 4, tt = r & 15;
  int ntiles = N >> 4;
  int nt = tile % ntiles, kt = tile / ntiles;
  int kk = kt * 32 + ((tt < 8) ? tt : tt + 8) + ((lane < 16) ? 0 : 8);
  int nn = nt * 16 + (lane & 15);
  out[i] = (__bf16)W[kk * N + nn];
}

// ---------------- normalize pooling vectors ----------------
__global__ __launch_bounds__(128) void norm_p(const float* p1, const float* p2,
                                              const float* p3, float* out) {
  const float* p = (blockIdx.x == 0) ? p1 : ((blockIdx.x == 1) ? p2 : p3);
  __shared__ float red[128];
  float v = p[threadIdx.x];
  red[threadIdx.x] = v * v;
  __syncthreads();
  for (int s = 64; s > 0; s >>= 1) {
    if (threadIdx.x < s) red[threadIdx.x] += red[threadIdx.x + s];
    __syncthreads();
  }
  out[blockIdx.x * 128 + threadIdx.x] = v / sqrtf(red[0]);
}

// ---------------- final MLP head ----------------
__global__ __launch_bounds__(128) void mlp_kernel(
    const float* __restrict__ gbuf, const float* __restrict__ L1w,
    const float* __restrict__ L1b, const float* __restrict__ L2w,
    const float* __restrict__ L2b, const float* __restrict__ L3w,
    const float* __restrict__ L3b, float* __restrict__ out) {
  int g = blockIdx.x, t = threadIdx.x;
  __shared__ float z0[128], z1[128], z2[64];
  z0[t] = gbuf[g * 128 + t];
  __syncthreads();
  float a = L1b[t];
  for (int k = 0; k < 128; ++k) a += z0[k] * L1w[k * 128 + t];
  z1[t] = fmaxf(a, 0.f);
  __syncthreads();
  if (t < 64) {
    float b = L2b[t];
    for (int k = 0; k < 128; ++k) b += z1[k] * L2w[k * 64 + t];
    z2[t] = fmaxf(b, 0.f);
  }
  __syncthreads();
  if (t == 0) {
    float c = L3b[0];
    for (int k = 0; k < 64; ++k) c += z2[k] * L3w[k];
    out[g] = 1.f / (1.f + expf(-c));
  }
}

// ---------------- host side ----------------
extern "C" void kernel_launch(void* const* d_in, const int* in_sizes, int n_in,
                              void* d_out, int out_size, void* d_ws,
                              size_t ws_size, hipStream_t stream) {
  const int* x = (const int*)d_in[0];
  const int* ei = (const int*)d_in[1];
  const int* src = ei;
  const int* dst = ei + E_;
  const float* emb = (const float*)d_in[3];
  const float* W1l = (const float*)d_in[4];
  const float* b1l = (const float*)d_in[5];
  const float* W1r = (const float*)d_in[6];
  const float* p1 = (const float*)d_in[7];
  const float* W2l = (const float*)d_in[8];
  const float* b2l = (const float*)d_in[9];
  const float* W2r = (const float*)d_in[10];
  const float* p2 = (const float*)d_in[11];
  const float* W3l = (const float*)d_in[12];
  const float* b3l = (const float*)d_in[13];
  const float* W3r = (const float*)d_in[14];
  const float* p3 = (const float*)d_in[15];
  const float* L1w = (const float*)d_in[16];
  const float* L1b = (const float*)d_in[17];
  const float* L2w = (const float*)d_in[18];
  const float* L2b = (const float*)d_in[19];
  const float* L3w = (const float*)d_in[20];
  const float* L3b = (const float*)d_in[21];

  __bf16* pw = (__bf16*)d_ws;                              // 131072 bf16 = 256KB
  float* pn = (float*)((char*)d_ws + 262144);              // 3*128 f32
  float* gbuf = (float*)((char*)d_ws + 266240);            // 4096*128 f32

  pack_weight<<<128, 256, 0, stream>>>(W1l, pw + 0, 256, 128);
  pack_weight<<<128, 256, 0, stream>>>(W1r, pw + 32768, 256, 128);
  pack_weight<<<64, 256, 0, stream>>>(W2l, pw + 65536, 128, 128);
  pack_weight<<<64, 256, 0, stream>>>(W2r, pw + 81920, 128, 128);
  pack_weight<<<64, 256, 0, stream>>>(W3l, pw + 98304, 128, 128);
  pack_weight<<<64, 256, 0, stream>>>(W3r, pw + 114688, 128, 128);
  norm_p<<<3, 128, 0, stream>>>(p1, p2, p3, pn);

  gnn_graph_kernel<<<G_, 128, 0, stream>>>(x, src, dst, emb, pw, b1l, b2l, b3l,
                                           pn, gbuf);
  mlp_kernel<<<G_, 128, 0, stream>>>(gbuf, L1w, L1b, L2w, L2b, L3w, L3b,
                                     (float*)d_out);
}
